// SparseStructureTransformer_8022998909238
// MI455X (gfx1250) — matrix operand order repaired and checked by reading, not compiled, not run on hardware
//
#include <hip/hip_runtime.h>

#define B_     8
#define N_     2048
#define EMBED_ 64
#define HID_   512
#define HEADS_ 8
#define LAYERS_ 6
#define DFF_   2048
#define KOUT_  16
#define CODES_ 8192

// padded LDS row strides (bf16 elems) to avoid 64-bank power-of-two conflicts;
// all are multiples of 8 so v8bf (16B) accesses stay aligned.
#define SK32   40   // tiles with 32 valid columns
#define SK64   72   // tiles with 64 valid columns

typedef __attribute__((ext_vector_type(16))) __bf16 v16bf;
typedef __attribute__((ext_vector_type(8)))  __bf16 v8bf;
typedef __attribute__((ext_vector_type(8)))  float  v8f;

__device__ __forceinline__ v8f wmma_bf16(v16bf a, v16bf b, v8f c) {
  // v_wmma_f32_16x16x32_bf16: D = A(16x32) * B(32x16) + C(16x16 f32)
  return __builtin_amdgcn_wmma_f32_16x16x32_bf16(false, a, false, b, (short)0, c, false, false);
}

// A fragment (16x32 bf16). Lane L holds row L&15; half = L>>4 selects K pattern:
// elems 0..7 -> K = half*8 + i ; elems 8..15 -> K = 16 + half*8 + (i-8).
__device__ __forceinline__ v16bf load_fragA(const __bf16* base, int row, int stride, int koff, int hf) {
  const v8bf* p0 = reinterpret_cast<const v8bf*>(base + row * stride + koff + hf * 8);
  const v8bf* p1 = reinterpret_cast<const v8bf*>(base + row * stride + koff + 16 + hf * 8);
  v8bf lo = *p0, hi = *p1;
  v16bf r;
#pragma unroll
  for (int i = 0; i < 8; ++i) { r[i] = lo[i]; r[i + 8] = hi[i]; }
  return r;
}

// B fragment (32x16 bf16). Lane L holds column L&15; same K pattern as A:
// elems 0..7 -> K = half*8 + i ; elems 8..15 -> K = 16 + half*8 + (i-8).
__device__ __forceinline__ v16bf load_fragB(const __bf16* base, int col, int stride, int koff, int hf) {
  const v8bf* p0 = reinterpret_cast<const v8bf*>(base + col * stride + koff + hf * 8);
  const v8bf* p1 = reinterpret_cast<const v8bf*>(base + col * stride + koff + 16 + hf * 8);
  v8bf lo = *p0, hi = *p1;
  v16bf r;
#pragma unroll
  for (int i = 0; i < 8; ++i) { r[i] = lo[i]; r[i + 8] = hi[i]; }
  return r;
}

__device__ __forceinline__ void cvt_store8(__bf16* dst, float4 a, float4 b) {
  v8bf v;
  v[0] = (__bf16)a.x; v[1] = (__bf16)a.y; v[2] = (__bf16)a.z; v[3] = (__bf16)a.w;
  v[4] = (__bf16)b.x; v[5] = (__bf16)b.y; v[6] = (__bf16)b.z; v[7] = (__bf16)b.w;
  *reinterpret_cast<v8bf*>(dst) = v;
}

// ---------------------------------------------------------------------------
// Fourier positional features + input projection (K=127, VALU kernel)
// ---------------------------------------------------------------------------
__global__ __launch_bounds__(128) void sst_inproj(
    const float* __restrict__ pos, const float* __restrict__ emb,
    const float* __restrict__ W, const float* __restrict__ bias,
    float* __restrict__ X)
{
  __shared__ float feat[128];
  int tok = blockIdx.x;
  int tid = threadIdx.x;
  if (tid < 3) {
    float n = pos[tok * 3 + tid] * (2.0f / 32.0f) - 1.0f;
    feat[tid] = n;
    float f = 3.14159265358979f;
#pragma unroll
    for (int j = 0; j < 10; ++j) {
      feat[3 + tid * 20 + j]      = __sinf(n * f);
      feat[3 + tid * 20 + 10 + j] = __cosf(n * f);
      f *= 2.0f;
    }
  }
  if (tid < 64) feat[63 + tid] = emb[(long)tok * 64 + tid];
  if (tid == 127) feat[127] = 0.0f;
  __syncthreads();
#pragma unroll
  for (int rr = 0; rr < 4; ++rr) {
    int r = tid + rr * 128;
    const float* w = W + (long)r * 127;
    float acc = bias[r];
    for (int k = 0; k < 127; ++k) acc = fmaf(feat[k], w[k], acc);
    X[(long)tok * 512 + r] = acc;
  }
}

// ---------------------------------------------------------------------------
// LayerNorm over rows of width 512 (256 threads, 2 elems each)
// ---------------------------------------------------------------------------
__global__ __launch_bounds__(256) void sst_ln(
    const float* __restrict__ X, const float* __restrict__ g,
    const float* __restrict__ b, float* __restrict__ Y)
{
  const int D = 512;
  long row = blockIdx.x;
  const float* x = X + row * D;
  float* y = Y + row * D;
  int tid = threadIdx.x;
  float v0 = x[tid], v1 = x[tid + 256];
  float s = v0 + v1, ss = v0 * v0 + v1 * v1;
#pragma unroll
  for (int off = 1; off < 32; off <<= 1) {
    s  += __shfl_xor(s, off, 32);
    ss += __shfl_xor(ss, off, 32);
  }
  __shared__ float sb[8], sq[8];
  int wid = tid >> 5, lane = tid & 31;
  if (lane == 0) { sb[wid] = s; sq[wid] = ss; }
  __syncthreads();
  if (tid == 0) {
    float ts = 0.f, tq = 0.f;
#pragma unroll
    for (int i = 0; i < 8; ++i) { ts += sb[i]; tq += sq[i]; }
    sb[0] = ts; sq[0] = tq;
  }
  __syncthreads();
  float mean = sb[0] * (1.0f / D);
  float var  = sq[0] * (1.0f / D) - mean * mean;
  float inv  = rsqrtf(var + 1e-5f);
  y[tid]       = (v0 - mean) * inv * g[tid] + b[tid];
  y[tid + 256] = (v1 - mean) * inv * g[tid + 256] + b[tid + 256];
}

// ---------------------------------------------------------------------------
// Zero-padded seed copy so the seeds GEMM needs no row guards (M=16 -> 64)
// ---------------------------------------------------------------------------
__global__ __launch_bounds__(256) void sst_seedpad(
    const float* __restrict__ seeds, float* __restrict__ dst)
{
  int i = blockIdx.x * 256 + threadIdx.x;     // 64*512 elems
  dst[i] = (i < KOUT_ * HID_) ? seeds[i] : 0.0f;
}

// ---------------------------------------------------------------------------
// Guard-free WMMA GEMM: C[M,N] = act(A[M,K] @ W[N,K]^T + bias (+ res))
// Requirements: M % 64 == 0, N % 128 == 0, K % 32 == 0 (all call sites comply).
// Block 256 thr = 8 waves; tile 64x128; wave computes 32x32 (4 WMMA / step);
// double-buffered LDS: next global tile loads overlap the WMMAs, one barrier
// per K-step.
// ---------------------------------------------------------------------------
__global__ __launch_bounds__(256) void sst_gemm(
    const float* __restrict__ A, const float* __restrict__ W,
    const float* __restrict__ bias, const float* __restrict__ res,
    float* __restrict__ C, int M, int N, int K, int relu)
{
  (void)M;
  __shared__ __bf16 As[2][64 * SK32];
  __shared__ __bf16 Ws[2][128 * SK32];
  int tid = threadIdx.x;
  int lane = tid & 31, wid = tid >> 5;
  int hf = lane >> 4, ln = lane & 15;
  int bm = blockIdx.x * 64, bn = blockIdx.y * 128;
  int wm = (wid >> 2) * 32;        // 0 / 32
  int wn = (wid & 3) * 32;         // 0,32,64,96
  int ar = tid >> 2, ac = (tid & 3) * 8;    // A staging: 8 elems/thread
  int wr = tid >> 1, wc = (tid & 1) * 16;   // W staging: 16 elems/thread
  const float* Ap = A + (long)(bm + ar) * K + ac;
  const float* Wp = W + (long)(bn + wr) * K + wc;

  float4 a0 = *reinterpret_cast<const float4*>(Ap);
  float4 a1 = *reinterpret_cast<const float4*>(Ap + 4);
  float4 w0 = *reinterpret_cast<const float4*>(Wp);
  float4 w1 = *reinterpret_cast<const float4*>(Wp + 4);
  float4 w2 = *reinterpret_cast<const float4*>(Wp + 8);
  float4 w3 = *reinterpret_cast<const float4*>(Wp + 12);
  cvt_store8(&As[0][ar * SK32 + ac], a0, a1);
  cvt_store8(&Ws[0][wr * SK32 + wc], w0, w1);
  cvt_store8(&Ws[0][wr * SK32 + wc + 8], w2, w3);
  __syncthreads();

  v8f acc00 = {0, 0, 0, 0, 0, 0, 0, 0};
  v8f acc01 = acc00, acc10 = acc00, acc11 = acc00;
  int buf = 0;
  for (int k0 = 0; k0 < K; k0 += 32) {
    bool more = (k0 + 32) < K;            // uniform branch
    if (more) {                            // loads in flight during WMMAs
      a0 = *reinterpret_cast<const float4*>(Ap + k0 + 32);
      a1 = *reinterpret_cast<const float4*>(Ap + k0 + 36);
      w0 = *reinterpret_cast<const float4*>(Wp + k0 + 32);
      w1 = *reinterpret_cast<const float4*>(Wp + k0 + 36);
      w2 = *reinterpret_cast<const float4*>(Wp + k0 + 40);
      w3 = *reinterpret_cast<const float4*>(Wp + k0 + 44);
    }
    v16bf fa0 = load_fragA(&As[buf][0], wm + ln, SK32, 0, hf);
    v16bf fa1 = load_fragA(&As[buf][0], wm + 16 + ln, SK32, 0, hf);
    v16bf fb0 = load_fragB(&Ws[buf][0], wn + ln, SK32, 0, hf);
    v16bf fb1 = load_fragB(&Ws[buf][0], wn + 16 + ln, SK32, 0, hf);
    acc00 = wmma_bf16(fa0, fb0, acc00);
    acc01 = wmma_bf16(fa0, fb1, acc01);
    acc10 = wmma_bf16(fa1, fb0, acc10);
    acc11 = wmma_bf16(fa1, fb1, acc11);
    if (more) {
      cvt_store8(&As[buf ^ 1][ar * SK32 + ac], a0, a1);
      cvt_store8(&Ws[buf ^ 1][wr * SK32 + wc], w0, w1);
      cvt_store8(&Ws[buf ^ 1][wr * SK32 + wc + 8], w2, w3);
    }
    __syncthreads();
    buf ^= 1;
  }

  int col0 = bn + wn + ln;
  int col1 = col0 + 16;
  float bv0 = bias ? bias[col0] : 0.0f;
  float bv1 = bias ? bias[col1] : 0.0f;
#pragma unroll
  for (int v = 0; v < 8; ++v) {
    int r0 = bm + wm + hf * 8 + v;        // C layout: row = v + half*8
    int r1 = r0 + 16;
    float x00 = acc00[v] + bv0, x01 = acc01[v] + bv1;
    float x10 = acc10[v] + bv0, x11 = acc11[v] + bv1;
    if (res) {
      x00 += res[(long)r0 * N + col0]; x01 += res[(long)r0 * N + col1];
      x10 += res[(long)r1 * N + col0]; x11 += res[(long)r1 * N + col1];
    }
    if (relu) {
      x00 = fmaxf(x00, 0.f); x01 = fmaxf(x01, 0.f);
      x10 = fmaxf(x10, 0.f); x11 = fmaxf(x11, 0.f);
    }
    C[(long)r0 * N + col0] = x00; C[(long)r0 * N + col1] = x01;
    C[(long)r1 * N + col0] = x10; C[(long)r1 * N + col1] = x11;
  }
}

// ---------------------------------------------------------------------------
// Flash attention, d=64 per head. Block = 128 thr (4 waves), 64 queries/block
// (16 per wave). Keys streamed in chunks of 32 with double-buffered LDS;
// QK^T and PV via bf16 WMMA; P converted C-layout -> A-layout through a
// per-wave LDS buffer (per-wave LDS is in-order).
// ---------------------------------------------------------------------------
__global__ __launch_bounds__(128) void sst_attn(
    const float* __restrict__ Q, long qBatch, int qStride,
    const float* __restrict__ Kp, long kBatch, int kStride,
    const float* __restrict__ Vp, long vBatch, int vStride,
    float* __restrict__ O, long oBatch, int oStride,
    int nQ, int nK, float scale)
{
  __shared__ __bf16 Qs[64 * SK64];        // queries x d
  __shared__ __bf16 Ks[2][32 * SK64];     // keys x d (double buffered)
  __shared__ __bf16 Vt[2][64 * SK32];     // d x keys (transposed, double buffered)
  __shared__ __bf16 Ps[4][16 * SK32];     // per-wave P tile
  int b = blockIdx.z, h = blockIdx.y;
  int qtile = blockIdx.x * 64;
  int tid = threadIdx.x, lane = tid & 31, wid = tid >> 5;
  int hf = lane >> 4, ln = lane & 15;
  const float* Qb = Q + (long)b * qBatch + h * 64;
  const float* Kb = Kp + (long)b * kBatch + h * 64;
  const float* Vb = Vp + (long)b * vBatch + h * 64;
  for (int i = tid; i < 64 * 64; i += 128) {
    int r = i >> 6, c = i & 63;
    int qr = qtile + r;
    float qv = (qr < nQ) ? Qb[(long)qr * qStride + c] : 0.0f;
    Qs[r * SK64 + c] = (__bf16)qv;
  }
  // stage key/value chunk 0
#pragma unroll
  for (int j = 0; j < 16; ++j) {
    int idx = tid + j * 128;
    int r = idx >> 6, c = idx & 63;
    Ks[0][r * SK64 + c] = (__bf16)Kb[(long)r * kStride + c];
    Vt[0][c * SK32 + r] = (__bf16)Vb[(long)r * vStride + c];
  }
  __syncthreads();
  v16bf qa0 = load_fragA(Qs, wid * 16 + ln, SK64, 0, hf);   // d 0..31
  v16bf qa1 = load_fragA(Qs, wid * 16 + ln, SK64, 32, hf);  // d 32..63
  v8f o0 = {0, 0, 0, 0, 0, 0, 0, 0}, o1 = o0, o2 = o0, o3 = o0;
  float m[8], lsum[8];
#pragma unroll
  for (int v = 0; v < 8; ++v) { m[v] = -1e30f; lsum[v] = 0.0f; }

  float kreg[16], vreg[16];
  int buf = 0;
  for (int kc = 0; kc < nK; kc += 32) {
    bool more = (kc + 32) < nK;           // uniform branch
    if (more) {                            // next chunk loads in flight
#pragma unroll
      for (int j = 0; j < 16; ++j) {
        int idx = tid + j * 128;
        int r = idx >> 6, c = idx & 63;
        kreg[j] = Kb[(long)(kc + 32 + r) * kStride + c];
        vreg[j] = Vb[(long)(kc + 32 + r) * vStride + c];
      }
    }
    // S = Q K^T (two 16-key tiles, each accumulated over d in 2 WMMAs)
    v16bf bk0a = load_fragB(&Ks[buf][0], ln, SK64, 0, hf);
    v16bf bk0b = load_fragB(&Ks[buf][0], ln, SK64, 32, hf);
    v16bf bk1a = load_fragB(&Ks[buf][0], 16 + ln, SK64, 0, hf);
    v16bf bk1b = load_fragB(&Ks[buf][0], 16 + ln, SK64, 32, hf);
    v8f s0 = {0, 0, 0, 0, 0, 0, 0, 0};
    v8f s1 = {0, 0, 0, 0, 0, 0, 0, 0};
    s0 = wmma_bf16(qa0, bk0a, s0);
    s0 = wmma_bf16(qa1, bk0b, s0);
    s1 = wmma_bf16(qa0, bk1a, s1);
    s1 = wmma_bf16(qa1, bk1b, s1);
    // online softmax; row (v + hf*8) is lane-local after 16-lane reductions
    float alpha[8];
#pragma unroll
    for (int v = 0; v < 8; ++v) {
      float e0 = s0[v] * scale;
      float e1 = s1[v] * scale;
      float mx = fmaxf(e0, e1);
#pragma unroll
      for (int off = 1; off < 16; off <<= 1) mx = fmaxf(mx, __shfl_xor(mx, off, 32));
      float mn = fmaxf(m[v], mx);
      alpha[v] = __expf(m[v] - mn);
      float p0 = __expf(e0 - mn);
      float p1 = __expf(e1 - mn);
      float rs = p0 + p1;
#pragma unroll
      for (int off = 1; off < 16; off <<= 1) rs += __shfl_xor(rs, off, 32);
      lsum[v] = lsum[v] * alpha[v] + rs;
      m[v] = mn;
      s0[v] = p0; s1[v] = p1;
    }
#pragma unroll
    for (int v = 0; v < 8; ++v) {
      o0[v] *= alpha[v]; o1[v] *= alpha[v]; o2[v] *= alpha[v]; o3[v] *= alpha[v];
    }
    // P: C-layout -> LDS -> A-layout (per-wave buffer)
    __bf16* P = &Ps[wid][0];
#pragma unroll
    for (int v = 0; v < 8; ++v) {
      P[(hf * 8 + v) * SK32 + ln]      = (__bf16)s0[v];
      P[(hf * 8 + v) * SK32 + 16 + ln] = (__bf16)s1[v];
    }
    __builtin_amdgcn_wave_barrier();
    v16bf pa  = load_fragA(P, ln, SK32, 0, hf);
    v16bf vb0 = load_fragB(&Vt[buf][0], ln, SK32, 0, hf);
    v16bf vb1 = load_fragB(&Vt[buf][0], 16 + ln, SK32, 0, hf);
    v16bf vb2 = load_fragB(&Vt[buf][0], 32 + ln, SK32, 0, hf);
    v16bf vb3 = load_fragB(&Vt[buf][0], 48 + ln, SK32, 0, hf);
    o0 = wmma_bf16(pa, vb0, o0);
    o1 = wmma_bf16(pa, vb1, o1);
    o2 = wmma_bf16(pa, vb2, o2);
    o3 = wmma_bf16(pa, vb3, o3);
    if (more) {
#pragma unroll
      for (int j = 0; j < 16; ++j) {
        int idx = tid + j * 128;
        int r = idx >> 6, c = idx & 63;
        Ks[buf ^ 1][r * SK64 + c] = (__bf16)kreg[j];
        Vt[buf ^ 1][c * SK32 + r] = (__bf16)vreg[j];
      }
    }
    __syncthreads();
    buf ^= 1;
  }
  float* Ob = O + (long)b * oBatch + h * 64;
#pragma unroll
  for (int v = 0; v < 8; ++v) {
    int row = qtile + wid * 16 + hf * 8 + v;
    if (row < nQ) {
      float inv = 1.0f / lsum[v];
      Ob[(long)row * oStride + ln]      = o0[v] * inv;
      Ob[(long)row * oStride + 16 + ln] = o1[v] * inv;
      Ob[(long)row * oStride + 32 + ln] = o2[v] * inv;
      Ob[(long)row * oStride + 48 + ln] = o3[v] * inv;
    }
  }
}

// ---------------------------------------------------------------------------
// VQ: per latent row, argmin over codes of (||c||^2 - 2 z.c); write z_q row
// and per-row squared error (deterministic reduction, no float atomics).
// ---------------------------------------------------------------------------
__global__ __launch_bounds__(256) void sst_vq(
    const float* __restrict__ latent, const float* __restrict__ cb,
    float* __restrict__ out, float* __restrict__ lpart)
{
  __shared__ float z[512];
  __shared__ float rbest[256];
  __shared__ int   ridx[256];
  int row = blockIdx.x, tid = threadIdx.x;
  z[tid]       = latent[(long)row * 512 + tid];
  z[tid + 256] = latent[(long)row * 512 + 256 + tid];
  __syncthreads();
  float best = 3.4e38f; int bidx = 0;
  for (int c = tid; c < CODES_; c += 256) {
    const float* cv = cb + (long)c * 512;
    if (c + 256 < CODES_)  // uniform per iteration -> global_prefetch_b8
      __builtin_prefetch(cb + (long)(c + 256) * 512, 0, 0);
    float dot = 0.f, n2 = 0.f;
    for (int k = 0; k < 512; k += 4) {
      float4 w  = *reinterpret_cast<const float4*>(cv + k);
      float4 zz = *reinterpret_cast<const float4*>(&z[k]);
      dot += zz.x * w.x + zz.y * w.y + zz.z * w.z + zz.w * w.w;
      n2  += w.x * w.x + w.y * w.y + w.z * w.z + w.w * w.w;
    }
    float d = n2 - 2.0f * dot;
    if (d < best) { best = d; bidx = c; }
  }
  rbest[tid] = best; ridx[tid] = bidx;
  __syncthreads();
  for (int s = 128; s > 0; s >>= 1) {
    if (tid < s) {
      float ob = rbest[tid + s]; int oi = ridx[tid + s];
      if (ob < rbest[tid] || (ob == rbest[tid] && oi < ridx[tid])) {
        rbest[tid] = ob; ridx[tid] = oi;
      }
    }
    __syncthreads();
  }
  int idx = ridx[0];
  const float* cv = cb + (long)idx * 512;
  float c0 = cv[tid], c1 = cv[tid + 256];
  out[(long)row * 512 + tid]       = c0;
  out[(long)row * 512 + 256 + tid] = c1;
  float d0 = z[tid] - c0, d1 = z[tid + 256] - c1;
  float part = d0 * d0 + d1 * d1;
#pragma unroll
  for (int off = 1; off < 32; off <<= 1) part += __shfl_xor(part, off, 32);
  __shared__ float pw[8];
  if ((tid & 31) == 0) pw[tid >> 5] = part;
  __syncthreads();
  if (tid == 0) {
    float t = 0.f;
#pragma unroll
    for (int i = 0; i < 8; ++i) t += pw[i];
    lpart[row] = t;
  }
}

__global__ __launch_bounds__(128) void sst_vqfin(
    const float* __restrict__ lpart, float* __restrict__ loss)
{
  __shared__ float s[128];
  int tid = threadIdx.x;
  s[tid] = lpart[tid];
  __syncthreads();
  for (int st = 64; st > 0; st >>= 1) {
    if (tid < st) s[tid] += s[tid + st];
    __syncthreads();
  }
  if (tid == 0) loss[0] = s[0] * (0.5f / 65536.0f);
}

// ---------------------------------------------------------------------------
extern "C" void kernel_launch(void* const* d_in, const int* in_sizes, int n_in,
                              void* d_out, int out_size, void* d_ws, size_t ws_size,
                              hipStream_t stream)
{
  (void)in_sizes; (void)n_in; (void)out_size; (void)ws_size;
  const float* positions   = (const float*)d_in[0];
  const float* embeddings  = (const float*)d_in[1];
  const float* inproj_w    = (const float*)d_in[2];
  const float* inproj_b    = (const float*)d_in[3];
  const float* enc_in_w    = (const float*)d_in[4];
  const float* enc_in_b    = (const float*)d_in[5];
  const float* enc_out_w   = (const float*)d_in[6];
  const float* enc_out_b   = (const float*)d_in[7];
  const float* ln1_g       = (const float*)d_in[8];
  const float* ln1_b       = (const float*)d_in[9];
  const float* ln2_g       = (const float*)d_in[10];
  const float* ln2_b       = (const float*)d_in[11];
  const float* ff1_w       = (const float*)d_in[12];
  const float* ff1_b       = (const float*)d_in[13];
  const float* ff2_w       = (const float*)d_in[14];
  const float* ff2_b       = (const float*)d_in[15];
  const float* seeds       = (const float*)d_in[16];
  const float* pool_in_w   = (const float*)d_in[17];
  const float* pool_in_b   = (const float*)d_in[18];
  const float* pool_out_w  = (const float*)d_in[19];
  const float* pool_out_b  = (const float*)d_in[20];
  const float* pool_proj_w = (const float*)d_in[21];
  const float* pool_proj_b = (const float*)d_in[22];
  const float* pool_ln_g   = (const float*)d_in[23];
  const float* pool_ln_b   = (const float*)d_in[24];
  const float* codebook    = (const float*)d_in[25];

  const int T = B_ * N_;  // 16384 tokens
  float* ws = (float*)d_ws;
  float* x       = ws;                           // 16384*512
  float* hbuf    = x + (long)T * HID_;           // 16384*512
  float* big     = hbuf + (long)T * HID_;        // 16384*2048 (qkv / ff1 / pool-kv)
  float* seedpad = big + (long)T * DFF_;         // 64*512 (zero-padded seeds)
  float* qp      = seedpad + 64 * HID_;          // 64*512 (rows 16..63 unused)
  float* opool   = qp + 64 * HID_;               // 128*512
  float* pooled  = opool + B_ * KOUT_ * HID_;    // 128*512
  float* latpre  = pooled + B_ * KOUT_ * HID_;   // 128*512
  float* latent  = latpre + B_ * KOUT_ * HID_;   // 128*512
  float* lpart   = latent + B_ * KOUT_ * HID_;   // 128

  sst_inproj<<<dim3(T), dim3(128), 0, stream>>>(positions, embeddings, inproj_w, inproj_b, x);

  for (int L = 0; L < LAYERS_; ++L) {
    sst_ln<<<dim3(T), dim3(256), 0, stream>>>(x, ln1_g + L * HID_, ln1_b + L * HID_, hbuf);
    sst_gemm<<<dim3(T / 64, (3 * HID_) / 128), dim3(256), 0, stream>>>(
        hbuf, enc_in_w + (long)L * 3 * HID_ * HID_, enc_in_b + (long)L * 3 * HID_,
        nullptr, big, T, 3 * HID_, HID_, 0);
    sst_attn<<<dim3(N_ / 64, HEADS_, B_), dim3(128), 0, stream>>>(
        big,            (long)N_ * 3 * HID_, 3 * HID_,
        big + HID_,     (long)N_ * 3 * HID_, 3 * HID_,
        big + 2 * HID_, (long)N_ * 3 * HID_, 3 * HID_,
        hbuf,           (long)N_ * HID_,     HID_,
        N_, N_, 0.125f);
    sst_gemm<<<dim3(T / 64, HID_ / 128), dim3(256), 0, stream>>>(
        hbuf, enc_out_w + (long)L * HID_ * HID_, enc_out_b + (long)L * HID_,
        x, x, T, HID_, HID_, 0);
    sst_ln<<<dim3(T), dim3(256), 0, stream>>>(x, ln2_g + L * HID_, ln2_b + L * HID_, hbuf);
    sst_gemm<<<dim3(T / 64, DFF_ / 128), dim3(256), 0, stream>>>(
        hbuf, ff1_w + (long)L * DFF_ * HID_, ff1_b + (long)L * DFF_,
        nullptr, big, T, DFF_, HID_, 1);
    sst_gemm<<<dim3(T / 64, HID_ / 128), dim3(256), 0, stream>>>(
        big, ff2_w + (long)L * HID_ * DFF_, ff2_b + (long)L * HID_,
        x, x, T, HID_, DFF_, 0);
  }

  // Pooling cross-attention: q = seeds@Wq^T (shared across batch), kv = x@Wkv^T
  sst_seedpad<<<dim3((64 * HID_) / 256), dim3(256), 0, stream>>>(seeds, seedpad);
  sst_gemm<<<dim3(1, HID_ / 128), dim3(256), 0, stream>>>(
      seedpad, pool_in_w, pool_in_b, nullptr, qp, 64, HID_, HID_, 0);
  sst_gemm<<<dim3(T / 64, (2 * HID_) / 128), dim3(256), 0, stream>>>(
      x, pool_in_w + (long)HID_ * HID_, pool_in_b + HID_,
      nullptr, big, T, 2 * HID_, HID_, 0);
  sst_attn<<<dim3(1, HEADS_, B_), dim3(128), 0, stream>>>(
      qp,         0L,                    HID_,
      big,        (long)N_ * 2 * HID_,   2 * HID_,
      big + HID_, (long)N_ * 2 * HID_,   2 * HID_,
      opool,      (long)KOUT_ * HID_,    HID_,
      KOUT_, N_, 0.125f);
  sst_gemm<<<dim3((B_ * KOUT_) / 64, HID_ / 128), dim3(256), 0, stream>>>(
      opool, pool_out_w, pool_out_b, nullptr, pooled, B_ * KOUT_, HID_, HID_, 0);
  sst_gemm<<<dim3((B_ * KOUT_) / 64, HID_ / 128), dim3(256), 0, stream>>>(
      pooled, pool_proj_w, pool_proj_b, nullptr, latpre, B_ * KOUT_, HID_, HID_, 0);
  sst_ln<<<dim3(B_ * KOUT_), dim3(256), 0, stream>>>(latpre, pool_ln_g, pool_ln_b, latent);

  float* out = (float*)d_out;
  sst_vq<<<dim3(B_ * KOUT_), dim3(256), 0, stream>>>(latent, codebook, out, lpart);
  sst_vqfin<<<dim3(1), dim3(128), 0, stream>>>(lpart, out + (long)B_ * KOUT_ * HID_);
}
